// QwenAttention_2027224563799
// MI455X (gfx1250) — compile-verified, matrix operand order checked by reading, not run on hardware
//
#include <hip/hip_runtime.h>
#include <hip/hip_bf16.h>
#include <cstdint>

typedef __attribute__((ext_vector_type(16))) __bf16 v16bf;
typedef __attribute__((ext_vector_type(8)))  float  v8f;
typedef __attribute__((ext_vector_type(4)))  int    v4i;
typedef __attribute__((ext_vector_type(4)))  float  v4f;

#define HIDN  4096
#define SEQL  2048
#define NHEAD 32
#define HDIM  128

// round-to-nearest-even fp32 -> bf16 (bits)
static __device__ __forceinline__ unsigned f2bf(float f) {
  unsigned u = __float_as_uint(f);
  return (u + 0x7FFFu + ((u >> 16) & 1u)) >> 16;
}
static __device__ __forceinline__ float bf2f(unsigned short h) {
  return __uint_as_float(((unsigned)h) << 16);
}
static __device__ __forceinline__ v4i pack8(v4f a, v4f b) {
  v4i p;
  p.x = f2bf(a.x) | (f2bf(a.y) << 16);
  p.y = f2bf(a.z) | (f2bf(a.w) << 16);
  p.z = f2bf(b.x) | (f2bf(b.y) << 16);
  p.w = f2bf(b.z) | (f2bf(b.w) << 16);
  return p;
}

union Frag { v4i q[2]; v16bf v; };

// ---------------------------------------------------------------------------
// C[M,N] = A[M,K] @ W[N,K]^T + bias   (M=SEQL, N=K=HIDN)
// Block tile 128x128, BK=32, 256 threads = 8 waves (4 M x 2 N),
// double-buffered LDS (global loads for tile k+1 overlap WMMAs on tile k),
// explicit global_prefetch for tile k+2. fp32 -> bf16 conversion at LDS store.
// ---------------------------------------------------------------------------
template <bool A_BF16, bool OUT_F32>
__global__ __launch_bounds__(256) void gemm_bias_k(
    const void* __restrict__ Ap, const float* __restrict__ W,
    const float* __restrict__ bias, void* __restrict__ Out) {
  constexpr int LDA = 40;  // bf16 row stride: 80B (16B-aligned, bank-spread)
  __shared__ alignas(16) unsigned short As[2][128 * LDA];
  __shared__ alignas(16) unsigned short Ws[2][128 * LDA];

  const int tid  = threadIdx.x;
  const int lane = tid & 31, wave = tid >> 5;
  const int wm = wave & 3, wn = wave >> 2;
  const int row0 = blockIdx.y * 128, col0 = blockIdx.x * 128;

  v8f acc[2][4];
#pragma unroll
  for (int i = 0; i < 2; i++)
#pragma unroll
    for (int j = 0; j < 4; j++)
      acc[i][j] = (v8f){0.f, 0.f, 0.f, 0.f, 0.f, 0.f, 0.f, 0.f};

  // cooperative loader: 128 rows x 32 cols, 2 threads/row, 16 elems each
  const int lr = tid >> 1;
  const int lc = (tid & 1) * 16;
  const size_t aOff = (size_t)(row0 + lr) * HIDN + lc;
  const size_t wOff = (size_t)(col0 + lr) * HIDN + lc;
  // fragment addressing (ISA 16-bit A/B layout)
  const int fr  = lane & 15;
  const int fkb = (lane < 16) ? 0 : 8;

  v4f fa[4], fw[4];
  v4i ra[2];

  auto load_regs = [&](int k0) {
    if (A_BF16) {
      const unsigned short* s = (const unsigned short*)Ap + aOff + k0;
      ra[0] = ((const v4i*)s)[0];
      ra[1] = ((const v4i*)s)[1];
    } else {
      const v4f* s = (const v4f*)((const float*)Ap + aOff + k0);
      fa[0] = s[0]; fa[1] = s[1]; fa[2] = s[2]; fa[3] = s[3];
    }
    const v4f* w = (const v4f*)(W + wOff + k0);
    fw[0] = w[0]; fw[1] = w[1]; fw[2] = w[2]; fw[3] = w[3];
  };
  auto store_lds = [&](int buf) {
    unsigned short* da = &As[buf][lr * LDA + lc];
    if (A_BF16) {
      ((v4i*)da)[0] = ra[0];
      ((v4i*)da)[1] = ra[1];
    } else {
      ((v4i*)da)[0] = pack8(fa[0], fa[1]);
      ((v4i*)da)[1] = pack8(fa[2], fa[3]);
    }
    unsigned short* dw = &Ws[buf][lr * LDA + lc];
    ((v4i*)dw)[0] = pack8(fw[0], fw[1]);
    ((v4i*)dw)[1] = pack8(fw[2], fw[3]);
  };

  load_regs(0);
  store_lds(0);
  __syncthreads();

  for (int k0 = 0; k0 < HIDN; k0 += 32) {
    const int cur = (k0 >> 5) & 1;
    const bool more = (k0 + 32) < HIDN;
    if (more) load_regs(k0 + 32);  // overlap with WMMAs below
    if (k0 + 64 < HIDN) {          // warm L2/L0 for the tile after that
      if (!A_BF16)
        __builtin_prefetch((const float*)Ap + aOff + k0 + 64, 0, 1);
      __builtin_prefetch(W + wOff + k0 + 64, 0, 1);
    }

    Frag af[2], bf_[4];
#pragma unroll
    for (int mt = 0; mt < 2; mt++) {
      const unsigned short* p = &As[cur][(wm * 32 + mt * 16 + fr) * LDA + fkb];
      af[mt].q[0] = *(const v4i*)p;
      af[mt].q[1] = *(const v4i*)(p + 16);
    }
#pragma unroll
    for (int nt = 0; nt < 4; nt++) {
      const unsigned short* p = &Ws[cur][(wn * 64 + nt * 16 + fr) * LDA + fkb];
      bf_[nt].q[0] = *(const v4i*)p;
      bf_[nt].q[1] = *(const v4i*)(p + 16);
    }
#pragma unroll
    for (int mt = 0; mt < 2; mt++)
#pragma unroll
      for (int nt = 0; nt < 4; nt++)
        acc[mt][nt] = __builtin_amdgcn_wmma_f32_16x16x32_bf16(
            false, af[mt].v, false, bf_[nt].v, (short)0, acc[mt][nt], false,
            false);

    if (more) store_lds(cur ^ 1);
    __syncthreads();
  }

  // epilogue: bias + store (C layout: lane->N, VGPR r -> M=r / r+8)
  const int rb = (lane < 16) ? 0 : 8;
#pragma unroll
  for (int mt = 0; mt < 2; mt++) {
    const int rbase = row0 + wm * 32 + mt * 16 + rb;
#pragma unroll
    for (int nt = 0; nt < 4; nt++) {
      const int col = col0 + wn * 64 + nt * 16 + (lane & 15);
      const float bv = bias[col];
#pragma unroll
      for (int rr = 0; rr < 8; rr++) {
        float val = acc[mt][nt][rr] + bv;
        size_t o = (size_t)(rbase + rr) * HIDN + col;
        if (OUT_F32)
          ((float*)Out)[o] = val;
        else
          ((unsigned short*)Out)[o] = (unsigned short)f2bf(val);
      }
    }
  }
}

// ---------------------------------------------------------------------------
// RoPE in-place on bf16 q,k. One thread per (pos, head, d<64) pair.
// ---------------------------------------------------------------------------
__global__ __launch_bounds__(256) void rope_k(unsigned short* __restrict__ q,
                                              unsigned short* __restrict__ k) {
  const int idx = blockIdx.x * 256 + threadIdx.x;  // SEQL*NHEAD*64 total
  const int d   = idx & 63;
  const int hh  = (idx >> 6) & (NHEAD - 1);
  const int pos = idx >> 11;
  const float inv = __expf(-(float)d * 0.14391157f);  // ln(10000)/64
  float s, c;
  __sincosf((float)pos * inv, &s, &c);
  const size_t o1 = (size_t)pos * HIDN + hh * HDIM + d;
  const size_t o2 = o1 + 64;
  {
    float x1 = bf2f(q[o1]), x2 = bf2f(q[o2]);
    q[o1] = (unsigned short)f2bf(x1 * c - x2 * s);
    q[o2] = (unsigned short)f2bf(x2 * c + x1 * s);
  }
  {
    float x1 = bf2f(k[o1]), x2 = bf2f(k[o2]);
    k[o1] = (unsigned short)f2bf(x1 * c - x2 * s);
    k[o2] = (unsigned short)f2bf(x2 * c + x1 * s);
  }
}

// ---------------------------------------------------------------------------
// Flash attention: block = (head, 128-query tile), 8 waves x 16 rows.
// K tile copied global->LDS with GLOBAL_LOAD_ASYNC_TO_LDS_B128 (ASYNCcnt);
// V tile stored transposed [128][32] so PV B-fragments are contiguous
// ds_load_b128. Online softmax, causal mask, scale = 1/ln(2048).
// ---------------------------------------------------------------------------
__global__ __launch_bounds__(256) void flash_attn_k(
    const unsigned short* __restrict__ Q, const unsigned short* __restrict__ K,
    const unsigned short* __restrict__ V, unsigned short* __restrict__ O) {
  constexpr int LDK = 136;  // 272B rows (16B-aligned)
  constexpr int LDV = 40;   // 80B rows (16B-aligned)
  __shared__ alignas(16) unsigned short Ks[32 * LDK];
  __shared__ alignas(16) unsigned short VTs[HDIM * LDV];
  __shared__ alignas(16) unsigned short Ps[8][16 * 32];

  const int head = blockIdx.x, qb = blockIdx.y;
  const int tid = threadIdx.x, lane = tid & 31, wave = tid >> 5;
  const int fr = lane & 15, fkb = (lane < 16) ? 0 : 8;
  const int rb = (lane < 16) ? 0 : 8;
  const int qrow0 = qb * 128 + wave * 16;
  const float SCALE = 0.13115409f;  // 1/ln(2048)

  // Q fragments: 16 rows x 128 (4 chunks of K=32), kept in registers
  Frag qf[4];
  {
    const unsigned short* qp =
        Q + (size_t)(qrow0 + fr) * HIDN + head * HDIM + fkb;
#pragma unroll
    for (int t = 0; t < 4; t++) {
      qf[t].q[0] = *(const v4i*)(qp + t * 32);
      qf[t].q[1] = *(const v4i*)(qp + t * 32 + 16);
    }
  }

  float m_[8], l_[8];
  v8f acc[8];
#pragma unroll
  for (int rr = 0; rr < 8; rr++) { m_[rr] = -INFINITY; l_[rr] = 0.f; }
#pragma unroll
  for (int dt = 0; dt < 8; dt++)
    acc[dt] = (v8f){0.f, 0.f, 0.f, 0.f, 0.f, 0.f, 0.f, 0.f};

  const int crow = tid >> 3;        // 0..31 kv rows
  const int ccol = (tid & 7) * 16;  // 0..112 d cols
  const int jend = qb * 128 + 128;  // causal horizon for this block

  // per-thread addresses for the async K-tile copy (32B per thread)
  const unsigned kLds = (unsigned)(uintptr_t)&Ks[crow * LDK + ccol];

  for (int j0 = 0; j0 < jend; j0 += 32) {
    {  // K tile row-major: async DMA global -> LDS (tracked by ASYNCcnt)
      unsigned long long ga = (unsigned long long)(uintptr_t)(
          K + (size_t)(j0 + crow) * HIDN + head * HDIM + ccol);
      asm volatile("global_load_async_to_lds_b128 %0, %1, off"
                   :: "v"(kLds), "v"(ga) : "memory");
      asm volatile("global_load_async_to_lds_b128 %0, %1, off offset:16"
                   :: "v"(kLds), "v"(ga) : "memory");
    }
    {  // V tile transposed at store time
      const unsigned short* s =
          V + (size_t)(j0 + crow) * HIDN + head * HDIM + ccol;
      unsigned short tmp[16];
      *(v4i*)&tmp[0] = *(const v4i*)s;
      *(v4i*)&tmp[8] = *(const v4i*)(s + 8);
#pragma unroll
      for (int e = 0; e < 16; e++) VTs[(ccol + e) * LDV + crow] = tmp[e];
    }
    asm volatile("s_wait_asynccnt 0" ::: "memory");
    __syncthreads();

    // scores for both 16-key subtiles (K-dim = HDIM in 4 steps of 32)
    v8f sc[2];
#pragma unroll
    for (int jj = 0; jj < 2; jj++) {
      sc[jj] = (v8f){0.f, 0.f, 0.f, 0.f, 0.f, 0.f, 0.f, 0.f};
      const unsigned short* kp = &Ks[(jj * 16 + fr) * LDK + fkb];
      Frag kf;
#pragma unroll
      for (int t = 0; t < 4; t++) {
        kf.q[0] = *(const v4i*)(kp + t * 32);
        kf.q[1] = *(const v4i*)(kp + t * 32 + 16);
        sc[jj] = __builtin_amdgcn_wmma_f32_16x16x32_bf16(
            false, qf[t].v, false, kf.v, (short)0, sc[jj], false, false);
      }
    }

    // single online-softmax update over all 32 keys of this step
#pragma unroll
    for (int rr = 0; rr < 8; rr++) {
      const int qrow = qrow0 + rb + rr;
      float x0 = sc[0][rr] * SCALE;
      float x1 = sc[1][rr] * SCALE;
      if (j0 + fr > qrow) x0 = -INFINITY;
      if (j0 + 16 + fr > qrow) x1 = -INFINITY;
      float rmax = fmaxf(x0, x1);
      rmax = fmaxf(rmax, __shfl_xor(rmax, 1, 32));
      rmax = fmaxf(rmax, __shfl_xor(rmax, 2, 32));
      rmax = fmaxf(rmax, __shfl_xor(rmax, 4, 32));
      rmax = fmaxf(rmax, __shfl_xor(rmax, 8, 32));
      const float nm = fmaxf(m_[rr], rmax);
      const float corr = (nm == -INFINITY) ? 1.0f : __expf(m_[rr] - nm);
      const float p0 = (x0 == -INFINITY) ? 0.f : __expf(x0 - nm);
      const float p1 = (x1 == -INFINITY) ? 0.f : __expf(x1 - nm);
      float rsum = p0 + p1;
      rsum += __shfl_xor(rsum, 1, 32);
      rsum += __shfl_xor(rsum, 2, 32);
      rsum += __shfl_xor(rsum, 4, 32);
      rsum += __shfl_xor(rsum, 8, 32);
      l_[rr] = l_[rr] * corr + rsum;
      m_[rr] = nm;
#pragma unroll
      for (int dt = 0; dt < 8; dt++) acc[dt][rr] *= corr;
      Ps[wave][(rb + rr) * 32 + fr] = (unsigned short)f2bf(p0);
      Ps[wave][(rb + rr) * 32 + 16 + fr] = (unsigned short)f2bf(p1);
    }
    __syncthreads();

    // O += P(16x32) x V(32x128)
    Frag pf;
    {
      const unsigned short* pp = &Ps[wave][fr * 32 + fkb];
      pf.q[0] = *(const v4i*)pp;
      pf.q[1] = *(const v4i*)(pp + 16);
    }
#pragma unroll
    for (int dt = 0; dt < 8; dt++) {
      const unsigned short* vp = &VTs[(dt * 16 + fr) * LDV + fkb];
      Frag vf;
      vf.q[0] = *(const v4i*)vp;
      vf.q[1] = *(const v4i*)(vp + 16);
      acc[dt] = __builtin_amdgcn_wmma_f32_16x16x32_bf16(
          false, pf.v, false, vf.v, (short)0, acc[dt], false, false);
    }
    __syncthreads();
  }

  // normalize + store bf16 attn output
#pragma unroll
  for (int rr = 0; rr < 8; rr++) {
    const float inv = 1.0f / l_[rr];
    const size_t row = (size_t)(qrow0 + rb + rr);
#pragma unroll
    for (int dt = 0; dt < 8; dt++)
      O[row * HIDN + head * HDIM + dt * 16 + fr] =
          (unsigned short)f2bf(acc[dt][rr] * inv);
  }
}

// ---------------------------------------------------------------------------
extern "C" void kernel_launch(void* const* d_in, const int* in_sizes, int n_in,
                              void* d_out, int out_size, void* d_ws,
                              size_t ws_size, hipStream_t stream) {
  const float* hs = (const float*)d_in[0];
  const float* Wq = (const float*)d_in[1];
  const float* bq = (const float*)d_in[2];
  const float* Wk = (const float*)d_in[3];
  const float* bk = (const float*)d_in[4];
  const float* Wv = (const float*)d_in[5];
  const float* bv = (const float*)d_in[6];
  const float* Wo = (const float*)d_in[7];
  const float* bo = (const float*)d_in[8];
  // d_in[9] = attention_mask (pure causal; applied analytically)

  unsigned short* qw = (unsigned short*)d_ws;
  unsigned short* kw = qw + (size_t)SEQL * HIDN;
  unsigned short* vw = kw + (size_t)SEQL * HIDN;
  unsigned short* aw = vw + (size_t)SEQL * HIDN;  // 64 MB total

  dim3 g(HIDN / 128, SEQL / 128), b(256);
  gemm_bias_k<false, false><<<g, b, 0, stream>>>(hs, Wq, bq, qw);
  gemm_bias_k<false, false><<<g, b, 0, stream>>>(hs, Wk, bk, kw);
  gemm_bias_k<false, false><<<g, b, 0, stream>>>(hs, Wv, bv, vw);
  rope_k<<<(SEQL * NHEAD * 64) / 256, 256, 0, stream>>>(qw, kw);
  flash_attn_k<<<dim3(NHEAD, SEQL / 128), b, 0, stream>>>(qw, kw, vw, aw);
  gemm_bias_k<true, true><<<g, b, 0, stream>>>(aw, Wo, bo, d_out);
}
